// RNNLM_31250182046138
// MI455X (gfx1250) — compile-verified, matrix operand order checked by reading, not run on hardware
//
#include <hip/hip_runtime.h>
#include <hip/hip_bf16.h>
#include <hip/hip_fp16.h>

// Problem sizes (fixed by the reference)
#define VOCAB  32000
#define EMBED  32
#define HIDDEN 16
#define SEQ    128
#define BATCH  64
#define NROWS  (SEQ * BATCH)   // 8192 global rows (s*64+b)
#define NCHUNK (VOCAB / 16)    // 2000 N-tiles of 16 vocab columns
#define NWAVES 8               // waves per output workgroup
#define CPW    (NCHUNK / NWAVES) // 250 chunks per wave (exact)

typedef __attribute__((ext_vector_type(16))) _Float16 v16h;
typedef __attribute__((ext_vector_type(8)))  float    v8f;
typedef __attribute__((ext_vector_type(4)))  int      i32x4;

// ---------------------------------------------------------------------------
// Kernel 1: h2o [16][32000] f32  ->  h2oT [32000][16] f16  (1 MB, L2 resident)
// 16x16 tile transpose through LDS; contiguous 512B writes per block.
// ---------------------------------------------------------------------------
__global__ __launch_bounds__(256) void k_transpose_h2o(const float* __restrict__ h2o,
                                                       _Float16* __restrict__ h2oT) {
    __shared__ _Float16 t[16][17];
    const int tid = threadIdx.x;
    const int x = tid & 15;   // n within tile (coalesced read dim)
    const int y = tid >> 4;   // k
    const int n0 = blockIdx.x * 16;
    float v = h2o[(size_t)y * VOCAB + (n0 + x)];
    t[x][y] = (_Float16)v;    // t[n_in_tile][k]
    __syncthreads();
    // thread (y=a, x=c) writes h2oT[(n0+a)*16 + c]; consecutive tid -> contiguous halves
    h2oT[(size_t)(n0 + y) * HIDDEN + x] = t[y][x];
}

// ---------------------------------------------------------------------------
// Kernel 2: P[g][j] = sum_e we[idx[g]][e] * i2h[e][j]   (g = s*64+b, 8192 rows)
// No sequential dependence -> fully parallel; removes random vocab gathers
// from the sequential recurrence.
// ---------------------------------------------------------------------------
__global__ __launch_bounds__(256) void k_embed_proj(const int* __restrict__ idx,
                                                    const float* __restrict__ we,
                                                    const float* __restrict__ i2h,
                                                    float* __restrict__ P) {
    __shared__ float wrow[16][EMBED];           // 16 gathered embedding rows
    __shared__ float wtop[EMBED][HIDDEN];       // i2h rows 0..31
    const int tid = threadIdx.x;
    const int gi = tid >> 4;                    // row within tile
    const int j  = tid & 15;                    // hidden column
    // load i2h top (512 f32) cooperatively
    for (int e = tid; e < EMBED * HIDDEN; e += 256)
        wtop[e >> 4][e & 15] = i2h[e];
    const int g  = blockIdx.x * 16 + gi;
    const int id = idx[g];
    wrow[gi][j]      = we[(size_t)id * EMBED + j];
    wrow[gi][j + 16] = we[(size_t)id * EMBED + j + 16];
    __syncthreads();
    float acc = 0.f;
#pragma unroll
    for (int e = 0; e < EMBED; ++e)
        acc += wrow[gi][e] * wtop[e][j];
    P[(size_t)g * HIDDEN + j] = acc;
}

// ---------------------------------------------------------------------------
// Kernel 3: sequential recurrence. One block, 1024 threads (=64x16, 32 waves).
// h_new[b][j] = tanh(P[s][b][j] + sum_k h[b][k]*i2h[32+k][j])
// Stores h as f16 rows [g][16] = exactly the WMMA A-operand feed layout.
// Software-pipelined prefetch of next step's P (contiguous 4KB/step).
// ---------------------------------------------------------------------------
__global__ __launch_bounds__(1024) void k_recurrent(const float* __restrict__ P,
                                                    const float* __restrict__ hidden0,
                                                    const float* __restrict__ i2h,
                                                    _Float16* __restrict__ h16) {
    __shared__ float hcur[BATCH][HIDDEN];
    __shared__ float wbot[HIDDEN][HIDDEN];      // i2h rows 32..47
    const int tid = threadIdx.x;
    const int b = tid >> 4;
    const int j = tid & 15;
    if (tid < HIDDEN * HIDDEN)
        wbot[tid >> 4][tid & 15] = i2h[(size_t)EMBED * HIDDEN + tid];
    hcur[b][j] = hidden0[tid];
    __syncthreads();
    float p = P[tid];
    for (int s = 0; s < SEQ; ++s) {
        float pn = (s + 1 < SEQ) ? P[(size_t)(s + 1) * (BATCH * HIDDEN) + tid] : 0.f;
        float acc = p;
#pragma unroll
        for (int k = 0; k < HIDDEN; ++k)
            acc += hcur[b][k] * wbot[k][j];
        float hn = tanhf(acc);
        __syncthreads();                         // all reads of hcur done
        hcur[b][j] = hn;
        h16[(size_t)s * (BATCH * HIDDEN) + tid] = (_Float16)hn;
        __syncthreads();                         // writes visible before next reads
        p = pn;
    }
}

// ---------------------------------------------------------------------------
// Kernel 4: vocab projection + log-softmax with V_WMMA_F32_16X16X32_F16.
// One 16-row tile per workgroup; 8 waves each own exactly 250 N-chunks
// (uniform, compile-time trip count => EXEC stays all-1s around every WMMA,
// no divergent-loop bookkeeping in the hot loop).
// Pass 1: accumulate sum(exp(logit)) per row (logits in [-3,5] => no max pass
// needed). Pass 2: recompute logits, write logit - log(Z) with NT stores.
// ---------------------------------------------------------------------------
__device__ __forceinline__ v8f tile_logits(v16h A, const _Float16* __restrict__ h2oT,
                                           int nbase, int m, int k0) {
    v16h B;
    i32x4 bv = *(const i32x4*)(h2oT + (size_t)(nbase + m) * HIDDEN + k0);
    i32x4 z = {0, 0, 0, 0};
    ((i32x4*)&B)[0] = bv;   // K = k0..k0+7 of column (nbase+m)
    ((i32x4*)&B)[1] = z;    // K = 16..31 zero-padded
    v8f c = {};
    c = __builtin_amdgcn_wmma_f32_16x16x32_f16(false, A, false, B,
                                               (short)0, c, false, false);
    return c;
}

__global__ __launch_bounds__(256) void k_output(const _Float16* __restrict__ h16,
                                                const _Float16* __restrict__ h2oT,
                                                float* __restrict__ out) {
    __shared__ float partials[NWAVES][16];
    const int tid  = threadIdx.x;
    const int lane = tid & 31;
    const int w    = __builtin_amdgcn_readfirstlane(tid >> 5); // wave id, scalar
    const int g0   = blockIdx.x * 16;           // first global row of tile
    const int m    = lane & 15;                 // row (A) / column (B) within tile
    const int k0   = (lane >> 4) << 3;          // K offset 0 or 8 per lane half
    const int rb   = (lane >> 4) << 3;          // C/D row base: lanes 16-31 hold M=8..15

    // A operand: constant across all N chunks (rows of h for this tile)
    v16h A;
    {
        i32x4 av = *(const i32x4*)(h16 + (size_t)(g0 + m) * HIDDEN + k0);
        i32x4 z = {0, 0, 0, 0};
        ((i32x4*)&A)[0] = av;
        ((i32x4*)&A)[1] = z;
    }

    // ---- Pass 1: per-row sum of exp(logits) ----
    float acc[8];
#pragma unroll
    for (int r = 0; r < 8; ++r) acc[r] = 0.f;
#pragma unroll 2
    for (int i = 0; i < CPW; ++i) {             // constant trip count: 250
        const int nbase = (i * NWAVES + w) * 16;
        v8f c = tile_logits(A, h2oT, nbase, m, k0);
#pragma unroll
        for (int r = 0; r < 8; ++r)
            acc[r] += __expf(c[r] + 1.0f);      // +1.0 = ones bias
    }
    // reduce over the 16-lane group (lanes 0-15 share rows 0-7; 16-31 rows 8-15)
#pragma unroll
    for (int r = 0; r < 8; ++r) {
        float v = acc[r];
        v += __shfl_xor(v, 1, 32);
        v += __shfl_xor(v, 2, 32);
        v += __shfl_xor(v, 4, 32);
        v += __shfl_xor(v, 8, 32);
        acc[r] = v;
    }
    if ((lane & 15) == 0) {
#pragma unroll
        for (int r = 0; r < 8; ++r)
            partials[w][rb + r] = acc[r];
    }
    __syncthreads();
    float logz[8];
#pragma unroll
    for (int r = 0; r < 8; ++r) {
        float t = 0.f;
#pragma unroll
        for (int ww = 0; ww < NWAVES; ++ww) t += partials[ww][rb + r];
        logz[r] = __logf(t);
    }

    // ---- Pass 2: recompute logits, write log-softmax with NT stores ----
#pragma unroll 2
    for (int i = 0; i < CPW; ++i) {             // constant trip count: 250
        const int nbase = (i * NWAVES + w) * 16;
        v8f c = tile_logits(A, h2oT, nbase, m, k0);
        const int n = nbase + m;
#pragma unroll
        for (int r = 0; r < 8; ++r) {
            float v = c[r] + 1.0f - logz[r];
            __builtin_nontemporal_store(v, out + (size_t)(g0 + rb + r) * VOCAB + n);
        }
    }
}

// ---------------------------------------------------------------------------
// Launch: inputs in setup_inputs() order: input(i32), hidden0, we, i2h, h2o.
// Workspace layout: [h2oT f16 1,024,000 B][h16 f16 262,144 B][P f32 524,288 B]
// ---------------------------------------------------------------------------
extern "C" void kernel_launch(void* const* d_in, const int* in_sizes, int n_in,
                              void* d_out, int out_size, void* d_ws, size_t ws_size,
                              hipStream_t stream) {
    const int*   idx     = (const int*)d_in[0];
    const float* hidden0 = (const float*)d_in[1];
    const float* we      = (const float*)d_in[2];
    const float* i2h     = (const float*)d_in[3];
    const float* h2o     = (const float*)d_in[4];
    float* out = (float*)d_out;

    char* ws = (char*)d_ws;
    _Float16* h2oT = (_Float16*)ws;
    _Float16* h16  = (_Float16*)(ws + (size_t)VOCAB * HIDDEN * 2);
    float*    P    = (float*)(ws + (size_t)VOCAB * HIDDEN * 2
                                 + (size_t)NROWS * HIDDEN * 2);

    k_transpose_h2o<<<NCHUNK,     256,  0, stream>>>(h2o, h2oT);
    k_embed_proj   <<<NROWS / 16, 256,  0, stream>>>(idx, we, i2h, P);
    k_recurrent    <<<1,          1024, 0, stream>>>(P, hidden0, i2h, h16);
    k_output       <<<NROWS / 16, 256,  0, stream>>>(h16, h2oT, out);
}
